// SwinBlock_18975165514154
// MI455X (gfx1250) — compile-verified, matrix operand order checked by reading, not run on hardware
//
#include <hip/hip_runtime.h>
#include <hip/hip_bf16.h>
#include <math.h>
#include <stdint.h>

// ---------------------------------------------------------------- types
typedef __attribute__((ext_vector_type(16))) __bf16 bf16x16;
typedef __attribute__((ext_vector_type(8)))  __bf16 bf16x8;
typedef __attribute__((ext_vector_type(2)))  __bf16 bf16x2;
typedef __attribute__((ext_vector_type(8)))  float  f32x8;

#define M_TOT   65536      // 16 * 4096 rows
#define CDIM    256
#define NWIN    1024       // 16 batches * 64 windows
#define QSCALE  0.17677669529663687f   // 32^-0.5

__device__ __forceinline__ __bf16 f2bf(float f) { return (__bf16)f; }

// WMMA D = A*B + C   (bf16 in, f32 accum), 16x16x32
__device__ __forceinline__ f32x8 wmma_bf16(bf16x16 a, bf16x16 b, f32x8 c) {
  return __builtin_amdgcn_wmma_f32_16x16x32_bf16(false, a, false, b,
                                                 (short)0, c, false, false);
}

// per-lane 16B memory -> LDS DMA, tracked by ASYNCcnt
__device__ __forceinline__ void async_load_b128(uint32_t lds_off, const void* gptr) {
  asm volatile("global_load_async_to_lds_b128 %0, %1, off"
               :: "v"(lds_off), "v"((uint64_t)(uintptr_t)gptr) : "memory");
}
__device__ __forceinline__ void wait_async0() {
  asm volatile("s_wait_asynccnt 0" ::: "memory");
}

// Build a 16-element A/B fragment from two contiguous 16B chunks:
// elements 0..7  <- p[0..7]   (K = g*8 .. g*8+7)
// elements 8..15 <- p[16..23] (K = 16+g*8 .. 23+g*8)
__device__ __forceinline__ bf16x16 frag2(const __bf16* p) {
  bf16x8 lo = *(const bf16x8*)p;
  bf16x8 hi = *(const bf16x8*)(p + 16);
  bf16x16 f;
#pragma unroll
  for (int i = 0; i < 8; ++i) { f[i] = lo[i]; f[i + 8] = hi[i]; }
  return f;
}

// window-order row -> original row (window reverse)
__device__ __forceinline__ int win_to_orig(int wr) {
  int t = wr & 63, win = wr >> 6;
  int b = win >> 6, wi = win & 63;
  int wy = wi >> 3, wx = wi & 7;
  int py = t >> 3,  px = t & 7;
  return b * 4096 + (wy * 8 + py) * 64 + (wx * 8 + px);
}

// ---------------------------------------------------------------- prep kernels
__global__ void wcast_t_kernel(const float* __restrict__ W, __bf16* __restrict__ Wt,
                               int K, int N) {
  int i = blockIdx.x * 256 + threadIdx.x;
  if (i >= K * N) return;
  int k = i / N, n = i % N;
  Wt[(size_t)n * K + k] = f2bf(W[i]);
}

__global__ void relbias_kernel(const float* __restrict__ table, float* __restrict__ biasM) {
  int i = blockIdx.x * 256 + threadIdx.x;      // 8*64*64
  if (i >= 8 * 64 * 64) return;
  int h = i >> 12, m = (i >> 6) & 63, n = i & 63;
  int my = m >> 3, mx = m & 7, ny = n >> 3, nx = n & 7;
  int idx = (my - ny + 7) * 15 + (mx - nx + 7);
  biasM[i] = table[idx * 8 + h];
}

// ---------------------------------------------------------------- layernorm
// one wave per 256-wide row; REORDER: 0 = keep order, 1 = window-partition
template<int REORDER>
__global__ __launch_bounds__(256) void ln_kernel(const float* __restrict__ x,
                                                 const float* __restrict__ gma,
                                                 const float* __restrict__ bta,
                                                 __bf16* __restrict__ out) {
  int row  = blockIdx.x * 8 + (threadIdx.x >> 5);
  int lane = threadIdx.x & 31;
  const float* xr = x + (size_t)row * CDIM + lane * 8;
  float v[8];
  float4 a0 = *(const float4*)xr, a1 = *(const float4*)(xr + 4);
  v[0]=a0.x; v[1]=a0.y; v[2]=a0.z; v[3]=a0.w;
  v[4]=a1.x; v[5]=a1.y; v[6]=a1.z; v[7]=a1.w;
  float s = 0.f, ss = 0.f;
#pragma unroll
  for (int j = 0; j < 8; ++j) { s += v[j]; ss += v[j] * v[j]; }
#pragma unroll
  for (int off = 1; off < 32; off <<= 1) {
    s  += __shfl_xor(s,  off, 32);
    ss += __shfl_xor(ss, off, 32);
  }
  float mu = s * (1.f / 256.f);
  float var = ss * (1.f / 256.f) - mu * mu;
  float rstd = rsqrtf(var + 1e-5f);
  int orow = REORDER ? 0 : row;
  if (REORDER) {
    int b = row >> 12, l = row & 4095;
    int y = l >> 6, xc = l & 63;
    int win = b * 64 + (y >> 3) * 8 + (xc >> 3);
    orow = win * 64 + (y & 7) * 8 + (xc & 7);
  }
  bf16x8 o;
#pragma unroll
  for (int j = 0; j < 8; ++j) {
    int c = lane * 8 + j;
    o[j] = f2bf((v[j] - mu) * rstd * gma[c] + bta[c]);
  }
  *(bf16x8*)(out + (size_t)orow * CDIM + lane * 8) = o;
}

// ---------------------------------------------------------------- GEMM
// C[M,N] = A[M,K] * Bt[N,K]^T ; block tile 128x128, 8 waves, wave tile 32x64.
// Double-buffered LDS, tiles staged with GLOBAL_LOAD_ASYNC_TO_LDS_B128 (ASYNCcnt),
// next tile's DMA issued before computing the current one.
// EPI: 0=qkv (scale q cols, bf16 out ld768)  1=proj (+bias +shortcut, f32 scatter ld256)
//      2=fc1 (+bias, gelu, bf16 ld1024)      3=fc2 (+bias +resid, f32 out ld256)
template<int EPI>
__global__ __launch_bounds__(256) void gemm_bf16_kernel(
    const __bf16* __restrict__ A, const __bf16* __restrict__ Bt,
    const float* __restrict__ ebias, const float* __restrict__ eres,
    void* __restrict__ out, int K) {
  __shared__ __bf16 sA[2][128 * 40];   // K-tile 32, row stride 40 (pad)
  __shared__ __bf16 sB[2][128 * 40];

  const int tid  = threadIdx.x;
  const int wv   = tid >> 5;
  const int lane = tid & 31;
  const int col  = lane & 15;
  const int g    = lane >> 4;
  const int wm   = wv & 3;          // 4 waves along M (32 rows each)
  const int wn   = wv >> 2;         // 2 waves along N (64 cols each)
  const int mB   = blockIdx.x * 128;
  const int nB   = blockIdx.y * 128;

  // staging: 256 threads x (2 A + 2 B) 16B chunks per 32-wide K tile
  const int srow = tid >> 2;            // 0..63
  const int sc   = (tid & 3) * 8;       // 0,8,16,24
  const __bf16* Ap = A  + (size_t)(mB + srow) * K + sc;
  const __bf16* Bp = Bt + (size_t)(nB + srow) * K + sc;
  const size_t rstep = (size_t)64 * K;

  // LDS byte offsets for this thread's chunks (buffer 0); buffer 1 = +10240
  const uint32_t la0 = (uint32_t)(uintptr_t)&sA[0][srow * 40 + sc];
  const uint32_t la1 = (uint32_t)(uintptr_t)&sA[0][(srow + 64) * 40 + sc];
  const uint32_t lb0 = (uint32_t)(uintptr_t)&sB[0][srow * 40 + sc];
  const uint32_t lb1 = (uint32_t)(uintptr_t)&sB[0][(srow + 64) * 40 + sc];
  const uint32_t BUFB = 128 * 40 * 2;   // bytes per buffer

  auto stage = [&](int kt, uint32_t bo) {
    async_load_b128(la0 + bo, Ap + kt);
    async_load_b128(la1 + bo, Ap + rstep + kt);
    async_load_b128(lb0 + bo, Bp + kt);
    async_load_b128(lb1 + bo, Bp + rstep + kt);
  };

  f32x8 acc[2][4] = {};
  auto compute = [&](const __bf16* bufA, const __bf16* bufB) {
    bf16x16 af[2], bfg[4];
#pragma unroll
    for (int mi = 0; mi < 2; ++mi)
      af[mi] = frag2(bufA + (wm * 32 + mi * 16 + col) * 40 + g * 8);
#pragma unroll
    for (int ni = 0; ni < 4; ++ni)
      bfg[ni] = frag2(bufB + (wn * 64 + ni * 16 + col) * 40 + g * 8);
#pragma unroll
    for (int mi = 0; mi < 2; ++mi)
#pragma unroll
      for (int ni = 0; ni < 4; ++ni)
        acc[mi][ni] = wmma_bf16(af[mi], bfg[ni], acc[mi][ni]);
  };

  stage(0, 0);
  for (int kt = 0; kt < K; kt += 64) {
    // ---- tile kt (buffer 0) ----
    wait_async0();
    __syncthreads();
    if (kt + 32 < K) stage(kt + 32, BUFB);
    if (kt + 64 < K) {
      __builtin_prefetch(Ap + kt + 64, 0, 1);
      __builtin_prefetch(Bp + kt + 64, 0, 1);
    }
    compute(sA[0], sB[0]);
    // ---- tile kt+32 (buffer 1) ----
    wait_async0();
    __syncthreads();
    if (kt + 64 < K) stage(kt + 64, 0);
    compute(sA[1], sB[1]);
  }

  // ---------------- epilogue ----------------
#pragma unroll
  for (int mi = 0; mi < 2; ++mi)
#pragma unroll
    for (int ni = 0; ni < 4; ++ni)
#pragma unroll
      for (int r = 0; r < 8; ++r) {
        int m = mB + wm * 32 + mi * 16 + r + 8 * g;
        int n = nB + wn * 64 + ni * 16 + col;
        float val = acc[mi][ni][r];
        if (EPI == 0) {                    // qkv: scale q (cols < 256)
          if (n < 256) val *= QSCALE;
          ((__bf16*)out)[(size_t)m * 768 + n] = f2bf(val);
        } else if (EPI == 1) {             // proj: +bias +shortcut, window-reverse
          int o = win_to_orig(m);
          ((float*)out)[(size_t)o * 256 + n] =
              val + ebias[n] + eres[(size_t)o * 256 + n];
        } else if (EPI == 2) {             // fc1: +bias, exact gelu
          float h = val + ebias[n];
          h = 0.5f * h * (1.0f + erff(h * 0.70710678118654752f));
          ((__bf16*)out)[(size_t)m * 1024 + n] = f2bf(h);
        } else {                           // fc2: +bias +residual -> final out
          ((float*)out)[(size_t)m * 256 + n] =
              val + ebias[n] + eres[(size_t)m * 256 + n];
        }
      }
}

// ---------------------------------------------------------------- attention
// one wave per (window, head). N=64 tokens, head_dim=32.
__global__ __launch_bounds__(32) void attn_kernel(const __bf16* __restrict__ qkv,
                                                  const float* __restrict__ biasM,
                                                  __bf16* __restrict__ out) {
  __shared__ __bf16 sAttn[64 * 72];   // softmaxed probabilities, padded rows
  __shared__ __bf16 sVT[32 * 72];     // v transposed [d][token], padded rows

  const int win  = blockIdx.x >> 3;
  const int h    = blockIdx.x & 7;
  const int lane = threadIdx.x;
  const int col  = lane & 15;
  const int g    = lane >> 4;

  const __bf16* qb = qkv + (size_t)(win * 64) * 768 + h * 32;        // q(t,d)
  const __bf16* kb = qb + 256;
  const __bf16* vb = qb + 512;

  // stage v transposed: sVT[d*72 + t]
  {
    int t0 = lane * 2;
#pragma unroll
    for (int c = 0; c < 4; ++c) {
      bf16x8 va = *(const bf16x8*)(vb + (size_t)t0 * 768 + c * 8);
      bf16x8 vc = *(const bf16x8*)(vb + (size_t)(t0 + 1) * 768 + c * 8);
#pragma unroll
      for (int j = 0; j < 8; ++j) {
        int d = c * 8 + j;
        bf16x2 p; p[0] = va[j]; p[1] = vc[j];
        *(bf16x2*)(sVT + d * 72 + t0) = p;
      }
    }
  }

  // scores S = q*scale @ k^T  (q pre-scaled in qkv epilogue)
  bf16x16 aq[4], bk[4];
#pragma unroll
  for (int mi = 0; mi < 4; ++mi)
    aq[mi] = frag2(qb + (size_t)(mi * 16 + col) * 768 + g * 8);
#pragma unroll
  for (int ni = 0; ni < 4; ++ni)
    bk[ni] = frag2(kb + (size_t)(ni * 16 + col) * 768 + g * 8);

  f32x8 s[4][4] = {};
#pragma unroll
  for (int mi = 0; mi < 4; ++mi)
#pragma unroll
    for (int ni = 0; ni < 4; ++ni)
      s[mi][ni] = wmma_bf16(aq[mi], bk[ni], s[mi][ni]);

  // + relative-position bias
  const float* bh = biasM + h * 4096;
#pragma unroll
  for (int mi = 0; mi < 4; ++mi)
#pragma unroll
    for (int ni = 0; ni < 4; ++ni)
#pragma unroll
      for (int r = 0; r < 8; ++r) {
        int m = mi * 16 + r + 8 * g;
        int n = ni * 16 + col;
        s[mi][ni][r] += bh[m * 64 + n];
      }

  // row softmax over 64 cols (4 ni fragments x 16 lanes of the half-wave)
#pragma unroll
  for (int mi = 0; mi < 4; ++mi)
#pragma unroll
    for (int r = 0; r < 8; ++r) {
      float mv = fmaxf(fmaxf(s[mi][0][r], s[mi][1][r]),
                       fmaxf(s[mi][2][r], s[mi][3][r]));
#pragma unroll
      for (int off = 1; off < 16; off <<= 1)
        mv = fmaxf(mv, __shfl_xor(mv, off, 32));
      float e0 = __expf(s[mi][0][r] - mv);
      float e1 = __expf(s[mi][1][r] - mv);
      float e2 = __expf(s[mi][2][r] - mv);
      float e3 = __expf(s[mi][3][r] - mv);
      float sm = e0 + e1 + e2 + e3;
#pragma unroll
      for (int off = 1; off < 16; off <<= 1)
        sm += __shfl_xor(sm, off, 32);
      float inv = 1.0f / sm;
      s[mi][0][r] = e0 * inv; s[mi][1][r] = e1 * inv;
      s[mi][2][r] = e2 * inv; s[mi][3][r] = e3 * inv;
    }

  // probabilities -> LDS (bf16)
#pragma unroll
  for (int mi = 0; mi < 4; ++mi)
#pragma unroll
    for (int ni = 0; ni < 4; ++ni)
#pragma unroll
      for (int r = 0; r < 8; ++r) {
        int m = mi * 16 + r + 8 * g;
        int n = ni * 16 + col;
        sAttn[m * 72 + n] = f2bf(s[mi][ni][r]);
      }
  asm volatile("s_wait_dscnt 0" ::: "memory");

  // O = P @ V   (K = 64 in two 32-steps)
  f32x8 o[4][2] = {};
#pragma unroll
  for (int kk = 0; kk < 2; ++kk) {
    bf16x16 ap[4], bv[2];
#pragma unroll
    for (int mi = 0; mi < 4; ++mi)
      ap[mi] = frag2(sAttn + (mi * 16 + col) * 72 + kk * 32 + g * 8);
#pragma unroll
    for (int ni = 0; ni < 2; ++ni)
      bv[ni] = frag2(sVT + (ni * 16 + col) * 72 + kk * 32 + g * 8);
#pragma unroll
    for (int mi = 0; mi < 4; ++mi)
#pragma unroll
      for (int ni = 0; ni < 2; ++ni)
        o[mi][ni] = wmma_bf16(ap[mi], bv[ni], o[mi][ni]);
  }

  __bf16* op = out + (size_t)(win * 64) * 256 + h * 32;
#pragma unroll
  for (int mi = 0; mi < 4; ++mi)
#pragma unroll
    for (int ni = 0; ni < 2; ++ni)
#pragma unroll
      for (int r = 0; r < 8; ++r) {
        int t = mi * 16 + r + 8 * g;
        int d = ni * 16 + col;
        op[(size_t)t * 256 + d] = f2bf(o[mi][ni][r]);
      }
}

// ---------------------------------------------------------------- host
extern "C" void kernel_launch(void* const* d_in, const int* in_sizes, int n_in,
                              void* d_out, int out_size, void* d_ws, size_t ws_size,
                              hipStream_t stream) {
  (void)in_sizes; (void)n_in; (void)out_size; (void)ws_size;
  const float* x      = (const float*)d_in[0];
  const float* ln1_g  = (const float*)d_in[3];
  const float* ln1_b  = (const float*)d_in[4];
  const float* qkv_w  = (const float*)d_in[5];
  const float* proj_w = (const float*)d_in[6];
  const float* proj_b = (const float*)d_in[7];
  const float* relt   = (const float*)d_in[8];
  const float* ln2_g  = (const float*)d_in[9];
  const float* ln2_b  = (const float*)d_in[10];
  const float* fc1_w  = (const float*)d_in[11];
  const float* fc1_b  = (const float*)d_in[12];
  const float* fc2_w  = (const float*)d_in[13];
  const float* fc2_b  = (const float*)d_in[14];

  char* ws = (char*)d_ws;
  size_t off = 0;
  auto alloc = [&](size_t bytes) {
    void* p = ws + off;
    off = (off + bytes + 255) & ~(size_t)255;
    return p;
  };
  __bf16* xw     = (__bf16*)alloc((size_t)M_TOT * 256 * 2);
  __bf16* qkvb   = (__bf16*)alloc((size_t)M_TOT * 768 * 2);
  __bf16* attn_o = (__bf16*)alloc((size_t)M_TOT * 256 * 2);
  float*  x2     = (float*) alloc((size_t)M_TOT * 256 * 4);
  __bf16* h2     = (__bf16*)alloc((size_t)M_TOT * 256 * 2);
  __bf16* fc1o   = (__bf16*)alloc((size_t)M_TOT * 1024 * 2);
  __bf16* WqkvT  = (__bf16*)alloc((size_t)768 * 256 * 2);
  __bf16* WprojT = (__bf16*)alloc((size_t)256 * 256 * 2);
  __bf16* Wfc1T  = (__bf16*)alloc((size_t)1024 * 256 * 2);
  __bf16* Wfc2T  = (__bf16*)alloc((size_t)256 * 1024 * 2);
  float*  biasM  = (float*) alloc((size_t)8 * 64 * 64 * 4);

  // weight transpose+cast, bias table expansion
  wcast_t_kernel<<<(256 * 768 + 255) / 256, 256, 0, stream>>>(qkv_w, WqkvT, 256, 768);
  wcast_t_kernel<<<(256 * 256 + 255) / 256, 256, 0, stream>>>(proj_w, WprojT, 256, 256);
  wcast_t_kernel<<<(256 * 1024 + 255) / 256, 256, 0, stream>>>(fc1_w, Wfc1T, 256, 1024);
  wcast_t_kernel<<<(1024 * 256 + 255) / 256, 256, 0, stream>>>(fc2_w, Wfc2T, 1024, 256);
  relbias_kernel<<<128, 256, 0, stream>>>(relt, biasM);

  // LN1 + window partition
  ln_kernel<1><<<M_TOT / 8, 256, 0, stream>>>(x, ln1_g, ln1_b, xw);
  // QKV projection (q pre-scaled)
  gemm_bf16_kernel<0><<<dim3(M_TOT / 128, 768 / 128), 256, 0, stream>>>(
      xw, WqkvT, nullptr, nullptr, qkvb, 256);
  // windowed attention
  attn_kernel<<<NWIN * 8, 32, 0, stream>>>(qkvb, biasM, attn_o);
  // proj + shortcut, window reverse -> x2 (original order)
  gemm_bf16_kernel<1><<<dim3(M_TOT / 128, 256 / 128), 256, 0, stream>>>(
      attn_o, WprojT, proj_b, x, x2, 256);
  // LN2
  ln_kernel<0><<<M_TOT / 8, 256, 0, stream>>>(x2, ln2_g, ln2_b, h2);
  // MLP
  gemm_bf16_kernel<2><<<dim3(M_TOT / 128, 1024 / 128), 256, 0, stream>>>(
      h2, Wfc1T, fc1_b, nullptr, fc1o, 256);
  gemm_bf16_kernel<3><<<dim3(M_TOT / 128, 256 / 128), 256, 0, stream>>>(
      fc1o, Wfc2T, fc2_b, x2, d_out, 1024);
}
